// RouterGate_62165356642908
// MI455X (gfx1250) — matrix operand order repaired and checked by reading, not compiled, not run on hardware
//
#include <hip/hip_runtime.h>
#include <hip/hip_bf16.h>
#include <math.h>

typedef __attribute__((ext_vector_type(16))) __bf16 v16bf;
typedef __attribute__((ext_vector_type(8)))  __bf16 v8bf;
typedef __attribute__((ext_vector_type(8)))  float  v8f;
typedef __attribute__((ext_vector_type(4)))  float  v4f;

#define D_DIM 2048
#define H_DIM 1024
#define E_DIM 64
#define N_TOK 16384

// ---------------------------------------------------------------------------
// One-shot f32 -> bf16 weight conversion (keeps WMMA inner loops convert-free)
// ---------------------------------------------------------------------------
__global__ __launch_bounds__(256) void k_cvt_f32_bf16(const float* __restrict__ s,
                                                      __bf16* __restrict__ d, int n) {
  int i = (blockIdx.x * 256 + threadIdx.x) * 4;
  if (i + 3 < n) {
    v4f v = *(const v4f*)(s + i);
    d[i + 0] = (__bf16)v.x;
    d[i + 1] = (__bf16)v.y;
    d[i + 2] = (__bf16)v.z;
    d[i + 3] = (__bf16)v.w;
  }
}

// ---------------------------------------------------------------------------
// Fused: h = x @ W1 + b1 ; LayerNorm(h) ; exact-erf GELU ; store bf16
// Block: 256 threads = 8 waves (2 M-groups x 4 N-groups), 32 rows x 1024 cols
// Each wave: 16 rows x 256 cols = 16 WMMA C-tiles (128 acc VGPRs)
// B fragments are software-pipelined (depth 2) so WMMA t overlaps the loads
// for tiles t+1 / t+2 instead of draining loadcnt to 0 per matrix op.
// ---------------------------------------------------------------------------
__global__ __launch_bounds__(256) void k_gemm1_ln_gelu(
    const float* __restrict__ x, const __bf16* __restrict__ w1b,
    const float* __restrict__ bias1, const float* __restrict__ gamma,
    const float* __restrict__ beta, __bf16* __restrict__ hact) {
  __shared__ __align__(16) __bf16 aT[32][40];  // 32 rows x 32 K (padded), bf16
  __shared__ float rsum[32], rsumsq[32];

  const int tid   = threadIdx.x;
  const int lane  = tid & 31;
  const int wid   = tid >> 5;
  const int waveM = wid >> 2;   // 0..1
  const int waveN = wid & 3;    // 0..3
  const int m0    = blockIdx.x * 32;
  const int r16   = lane & 15;
  const int half  = lane >> 4;  // 0 or 1

  if (tid < 32) { rsum[tid] = 0.0f; rsumsq[tid] = 0.0f; }

  v8f acc[16] = {};

  const int ldr = tid >> 3;        // staging row 0..31
  const int ldc = (tid & 7) * 4;   // staging col 0..28

  for (int k0 = 0; k0 < D_DIM; k0 += 32) {
    __syncthreads();
    {  // stage x tile [32 x 32] fp32 -> bf16 into LDS
      const float* p = x + (size_t)(m0 + ldr) * D_DIM + (k0 + ldc);
      v4f v = *(const v4f*)p;
      aT[ldr][ldc + 0] = (__bf16)v.x;
      aT[ldr][ldc + 1] = (__bf16)v.y;
      aT[ldr][ldc + 2] = (__bf16)v.z;
      aT[ldr][ldc + 3] = (__bf16)v.w;
      __builtin_prefetch(p + 32, 0, 1);  // next K tile -> global_prefetch_b8
    }
    __syncthreads();

    // A fragment (ISA 16-bit A 16x32 layout): two 8-K chunks at koff, koff+16
    const int arow = waveM * 16 + r16;
    const int koff = half * 8;
    v8bf a0 = *(const v8bf*)&aT[arow][koff];
    v8bf a1 = *(const v8bf*)&aT[arow][16 + koff];
    v16bf a = __builtin_shufflevector(a0, a1, 0, 1, 2, 3, 4, 5, 6, 7,
                                      8, 9, 10, 11, 12, 13, 14, 15);

    // B fragments: lane L holds W1 col r16 of each tile, K rows in A's table order
    const __bf16* wr = w1b + (size_t)(k0 + half * 8) * H_DIM + waveN * 256 + r16;
    auto ldB = [&](int tt) { v16bf r; for (int s = 0; s < 16; ++s) r[s] = wr[(size_t)(s + (s >= 8 ? 8 : 0)) * H_DIM + tt * 16]; return r; };
    __builtin_prefetch(wr + (size_t)32 * H_DIM, 0, 1);

    v16bf bbuf[2];
    bbuf[0] = ldB(0);
    bbuf[1] = ldB(1);
#pragma unroll
    for (int t = 0; t < 16; ++t) {
      const v16bf bc = bbuf[t & 1];
      if (t + 2 < 16) bbuf[t & 1] = ldB(t + 2);
      acc[t] = __builtin_amdgcn_wmma_f32_16x16x32_bf16(
          false, a, false, bc, (short)0, acc[t], false, false);
    }
  }

  // epilogue: + b1
#pragma unroll
  for (int t = 0; t < 16; ++t) {
    const int col = waveN * 256 + t * 16 + r16;
    const float bb = bias1[col];
#pragma unroll
    for (int v = 0; v < 8; ++v) acc[t][v] += bb;
  }

  // LayerNorm row reduction across the 4 N-waves via LDS float atomics
#pragma unroll
  for (int v = 0; v < 8; ++v) {
    const int rl = waveM * 16 + half * 8 + v;  // row within block (0..31)
    float s = 0.0f, ss = 0.0f;
#pragma unroll
    for (int t = 0; t < 16; ++t) {
      const float h = acc[t][v];
      s += h;
      ss += h * h;
    }
    atomicAdd(&rsum[rl], s);
    atomicAdd(&rsumsq[rl], ss);
  }
  __syncthreads();

  // normalize, exact-erf GELU, store bf16 activations
#pragma unroll
  for (int v = 0; v < 8; ++v) {
    const int rl   = waveM * 16 + half * 8 + v;
    const float mu  = rsum[rl] * (1.0f / H_DIM);
    const float var = rsumsq[rl] * (1.0f / H_DIM) - mu * mu;
    const float rs  = rsqrtf(var + 1e-5f);
    __bf16* orow = hact + (size_t)(m0 + rl) * H_DIM;
#pragma unroll
    for (int t = 0; t < 16; ++t) {
      const int col = waveN * 256 + t * 16 + r16;
      float hv = (acc[t][v] - mu) * rs * gamma[col] + beta[col];
      hv = 0.5f * hv * (1.0f + erff(hv * 0.70710678118654752f));
      orow[col] = (__bf16)hv;
    }
  }
}

// ---------------------------------------------------------------------------
// Fused: logits = h_act @ W2 + b2 ; softmax ; top-2 ; renormalize
// Block: 256 threads = 8 waves, each wave handles 16 token rows x 64 experts
// ---------------------------------------------------------------------------
__global__ __launch_bounds__(256) void k_gemm2_softmax_topk(
    const __bf16* __restrict__ hact, const __bf16* __restrict__ w2b,
    const float* __restrict__ b2, float* __restrict__ oidx,
    float* __restrict__ ow, float* __restrict__ olog) {
  __shared__ float lg[8][16][68];  // padded to dodge bank conflicts

  const int tid  = threadIdx.x;
  const int lane = tid & 31;
  const int wid  = tid >> 5;
  const int r16  = lane & 15;
  const int half = lane >> 4;
  const size_t r0 = (size_t)blockIdx.x * 128 + (size_t)wid * 16;

  v8f acc[4] = {};
  for (int k0 = 0; k0 < H_DIM; k0 += 32) {
    const __bf16* hr = hact + (r0 + r16) * H_DIM + k0 + half * 8;
    v8bf a0 = *(const v8bf*)hr;
    v8bf a1 = *(const v8bf*)(hr + 16);
    v16bf a = __builtin_shufflevector(a0, a1, 0, 1, 2, 3, 4, 5, 6, 7,
                                      8, 9, 10, 11, 12, 13, 14, 15);
    const __bf16* wr = w2b + (size_t)(k0 + half * 8) * E_DIM + r16;
    auto ldB = [&](int tt) { v16bf r; for (int s = 0; s < 16; ++s) r[s] = wr[(size_t)(s + (s >= 8 ? 8 : 0)) * E_DIM + tt * 16]; return r; };

    v16bf bbuf[2];
    bbuf[0] = ldB(0);
    bbuf[1] = ldB(1);
#pragma unroll
    for (int t = 0; t < 4; ++t) {
      const v16bf bc = bbuf[t & 1];
      if (t + 2 < 4) bbuf[t & 1] = ldB(t + 2);
      acc[t] = __builtin_amdgcn_wmma_f32_16x16x32_bf16(
          false, a, false, bc, (short)0, acc[t], false, false);
    }
  }

  // bias, emit logits (fp32) to global + LDS for the softmax stage
#pragma unroll
  for (int t = 0; t < 4; ++t) {
    const int col = t * 16 + r16;
    const float bb = b2[col];
#pragma unroll
    for (int v = 0; v < 8; ++v) {
      const int m  = half * 8 + v;
      const float L = acc[t][v] + bb;
      olog[(r0 + m) * E_DIM + col] = L;
      lg[wid][m][col] = L;
    }
  }
  __syncthreads();

  if (lane < 16) {  // one lane per token row
    const float* p = lg[wid][lane];
    float mx = -1e30f;
    for (int e = 0; e < 64; ++e) mx = fmaxf(mx, p[e]);
    float s = 0.0f;
    for (int e = 0; e < 64; ++e) s += expf(p[e] - mx);
    float v0 = -1e30f, v1 = -1e30f;
    int i0 = 0, i1 = 0;
    for (int e = 0; e < 64; ++e) {
      const float xv = p[e];
      if (xv > v0) { v1 = v0; i1 = i0; v0 = xv; i0 = e; }
      else if (xv > v1) { v1 = xv; i1 = e; }
    }
    const float p0 = expf(v0 - mx) / s;
    const float p1 = expf(v1 - mx) / s;
    const float inv = 1.0f / (p0 + p1 + 1e-9f);
    const size_t row = r0 + lane;
    oidx[row * 2 + 0] = (float)i0;
    oidx[row * 2 + 1] = (float)i1;
    ow[row * 2 + 0] = p0 * inv;
    ow[row * 2 + 1] = p1 * inv;
  }
}

// ---------------------------------------------------------------------------
extern "C" void kernel_launch(void* const* d_in, const int* in_sizes, int n_in,
                              void* d_out, int out_size, void* d_ws, size_t ws_size,
                              hipStream_t stream) {
  const float* x     = (const float*)d_in[0];
  const float* W1    = (const float*)d_in[1];
  const float* b1    = (const float*)d_in[2];
  const float* gamma = (const float*)d_in[3];
  const float* beta  = (const float*)d_in[4];
  const float* W2    = (const float*)d_in[5];
  const float* b2    = (const float*)d_in[6];

  float* out  = (float*)d_out;
  float* oidx = out;                      // [B,S,K] as float
  float* ow   = out + (size_t)N_TOK * 2;  // [B,S,K]
  float* olog = out + (size_t)N_TOK * 4;  // [N,E]

  char* ws = (char*)d_ws;
  __bf16* hact = (__bf16*)ws;                                            // 32 MB
  __bf16* w1b  = (__bf16*)(ws + (size_t)N_TOK * H_DIM * 2);              // 4 MB
  __bf16* w2b  = (__bf16*)(ws + (size_t)N_TOK * H_DIM * 2 +
                           (size_t)D_DIM * H_DIM * 2);                   // 128 KB

  k_cvt_f32_bf16<<<(D_DIM * H_DIM) / 1024, 256, 0, stream>>>(W1, w1b, D_DIM * H_DIM);
  k_cvt_f32_bf16<<<(H_DIM * E_DIM) / 1024, 256, 0, stream>>>(W2, w2b, H_DIM * E_DIM);
  k_gemm1_ln_gelu<<<N_TOK / 32, 256, 0, stream>>>(x, w1b, b1, gamma, beta, hact);
  k_gemm2_softmax_topk<<<N_TOK / 128, 256, 0, stream>>>(hact, w2b, b2, oidx, ow, olog);
}
